// QuantLlamaAttention_14207751815300
// MI455X (gfx1250) — matrix operand order repaired and checked by reading, not compiled, not run on hardware
//
#include <hip/hip_runtime.h>
#include <hip/hip_bf16.h>
#include <math.h>

// ---------------------------------------------------------------------------
// QuantLlamaAttention for gfx1250 (MI455X), wave32 + v_wmma_f32_16x16x32_f16
// + GLOBAL_LOAD_ASYNC_TO_LDS_B128 double-buffered staging (ASYNCcnt pipelined)
// B=1, S=2048, HID=4096, NH=32, NKV=8, HD=128, GQA groups=4
// ---------------------------------------------------------------------------

typedef _Float16 f16;
typedef __attribute__((ext_vector_type(16))) _Float16 v16h;
typedef __attribute__((ext_vector_type(8)))  float    v8f;
typedef int v4i_vs __attribute__((vector_size(16)));   // matches builtin param

#define DEV static __device__ __forceinline__
#define AS1 __attribute__((address_space(1)))
#define AS3 __attribute__((address_space(3)))

#if __has_builtin(__builtin_amdgcn_global_load_async_to_lds_b128) && \
    __has_builtin(__builtin_amdgcn_s_wait_asynccnt)
#define USE_ASYNC_COPY 1
#else
#define USE_ASYNC_COPY 0
#endif

constexpr int S    = 2048;
constexpr int HID  = 4096;
constexpr int NH   = 32;
constexpr int NKV  = 8;
constexpr int HD   = 128;
constexpr int KVW  = NKV * HD;          // 1024
constexpr float SM_SCALE = 0.08838834764831845f;  // 1/sqrt(128)

DEV v16h ld_frag(const f16* p0, const f16* p1) {
    union { v16h v; uint4 u[2]; } t;
    t.u[0] = *(const uint4*)p0;
    t.u[1] = *(const uint4*)p1;
    return t.v;
}

DEV v8f wmma_f16(v16h a, v16h b, v8f c) {
    // D = A(16x32 f16) * B(32x16 f16) + C(16x16 f32)
    return __builtin_amdgcn_wmma_f32_16x16x32_f16(false, a, false, b,
                                                  (short)0, c, false, false);
}

DEV float q127(float x, float sc) {
    float r = rintf(x / sc);
    r = fminf(fmaxf(r, -128.0f), 127.0f);
    return r * sc;
}

#if USE_ASYNC_COPY
// async copy of 16 f16 (2 x b128) from global directly into LDS
DEV void async_cp16(const f16* g, f16* l) {
    __builtin_amdgcn_global_load_async_to_lds_b128(
        (AS1 v4i_vs*)g, (AS3 v4i_vs*)l, 0, 0);
    __builtin_amdgcn_global_load_async_to_lds_b128(
        (AS1 v4i_vs*)(g + 8), (AS3 v4i_vs*)(l + 8), 0, 0);
}
#endif

// ---------------------------------------------------------------------------
// 1) hs = fq_per_token(hidden - qbias1) + qbias1     -> f16 [S, HID]
// ---------------------------------------------------------------------------
__global__ __launch_bounds__(256)
void hs_prep_kernel(const float* __restrict__ hidden,
                    const float* __restrict__ qb1,
                    f16* __restrict__ hs_h) {
    __shared__ float red[256];
    int s = blockIdx.x, tid = threadIdx.x;
    const float* row = hidden + (size_t)s * HID;
    float xv[16];
    float m = 0.0f;
#pragma unroll
    for (int i = 0; i < 16; ++i) {
        int j = tid + 256 * i;
        float x = row[j] - qb1[j];
        xv[i] = x;
        m = fmaxf(m, fabsf(x));
    }
    red[tid] = m; __syncthreads();
    for (int st = 128; st > 0; st >>= 1) {
        if (tid < st) red[tid] = fmaxf(red[tid], red[tid + st]);
        __syncthreads();
    }
    float sc = fmaxf(red[0], 1e-8f) / 127.0f;
    f16* orow = hs_h + (size_t)s * HID;
#pragma unroll
    for (int i = 0; i < 16; ++i) {
        int j = tid + 256 * i;
        orow[j] = (f16)(q127(xv[i], sc) + qb1[j]);
    }
}

// ---------------------------------------------------------------------------
// 2) weight f32 -> f16
// ---------------------------------------------------------------------------
__global__ __launch_bounds__(256)
void f32_to_f16_kernel(const float* __restrict__ w, f16* __restrict__ wh, int n) {
    int i = blockIdx.x * 256 + threadIdx.x;
    if (i < n) wh[i] = (f16)w[i];
}

// ---------------------------------------------------------------------------
// 3) NT GEMM:  C[M,N] = A[M,K] * W[N,K]^T   (A,W f16 row-major, C f32)
//    128x128x32 tiles, 8 waves, each wave -> 32x64 (2x4 WMMA tiles)
//    Async double-buffered LDS staging when available.
// ---------------------------------------------------------------------------
__global__ __launch_bounds__(256)
void gemm_nt_f16_kernel(const f16* __restrict__ A, const f16* __restrict__ W,
                        float* __restrict__ C, int M, int N, int K) {
#if USE_ASYNC_COPY
    __shared__ f16 As[2][128 * 40];   // padded stride 40 halves
    __shared__ f16 Ws[2][128 * 40];
#else
    __shared__ f16 As1[128 * 40];
    __shared__ f16 Ws1[128 * 40];
#endif

    int nb = N >> 7;
    int bm = blockIdx.x / nb, bn = blockIdx.x % nb;
    int tid = threadIdx.x, lane = tid & 31, wave = tid >> 5;
    int wm = wave & 3, wn = wave >> 2;      // 4 x 2 wave grid
    int ml = lane & 15, hf = lane >> 4;

    v8f acc[2][4];
#pragma unroll
    for (int i = 0; i < 2; ++i)
#pragma unroll
        for (int j = 0; j < 4; ++j)
#pragma unroll
            for (int v = 0; v < 8; ++v) acc[i][j][v] = 0.0f;

    int row = tid >> 1, seg = tid & 1;
    const f16* ga0 = A + (size_t)(bm * 128 + row) * K + seg * 16;
    const f16* gw0 = W + (size_t)(bn * 128 + row) * K + seg * 16;
    int loff = row * 40 + seg * 16;
    int nk = K >> 5;

#if USE_ASYNC_COPY
    // prologue: tile 0 -> buffer 0
    async_cp16(ga0, &As[0][loff]);
    async_cp16(gw0, &Ws[0][loff]);

    for (int i = 0; i < nk; ++i) {
        int cur = i & 1, nxt = cur ^ 1;
        if (i + 1 < nk) {
            const f16* ga = ga0 + (i + 1) * 32;
            const f16* gw = gw0 + (i + 1) * 32;
            async_cp16(ga, &As[nxt][loff]);     // tile i+1 streams in while
            async_cp16(gw, &Ws[nxt][loff]);     // tile i is being computed
            __builtin_amdgcn_s_wait_asynccnt(4); // tile i has landed
        } else {
            __builtin_amdgcn_s_wait_asynccnt(0);
        }
        __syncthreads();

        const f16* Ab = As[cur];
        const f16* Wb = Ws[cur];
        v16h af[2], bf[4];
#pragma unroll
        for (int mt = 0; mt < 2; ++mt) {
            const f16* p = &Ab[(wm * 32 + mt * 16 + ml) * 40];
            af[mt] = ld_frag(p + hf * 8, p + 16 + hf * 8);
        }
#pragma unroll
        for (int nt = 0; nt < 4; ++nt) {
            const f16* p = &Wb[(wn * 64 + nt * 16 + ml) * 40];
            bf[nt] = ld_frag(p + hf * 8, p + 16 + hf * 8);
        }
#pragma unroll
        for (int mt = 0; mt < 2; ++mt)
#pragma unroll
            for (int nt = 0; nt < 4; ++nt)
                acc[mt][nt] = wmma_f16(af[mt], bf[nt], acc[mt][nt]);
        __syncthreads();   // all waves done reading buf cur before it is refilled
    }
#else
    f16* la = &As1[loff];
    f16* lw = &Ws1[loff];
    for (int kb = 0; kb < K; kb += 32) {
        const f16* ga = ga0 + kb;
        const f16* gw = gw0 + kb;
        // batch all global loads, then one wait, then LDS stores
        uint4 t0 = *(const uint4*)ga;
        uint4 t1 = *(const uint4*)(ga + 8);
        uint4 t2 = *(const uint4*)gw;
        uint4 t3 = *(const uint4*)(gw + 8);
        if (kb + 32 < K) {
            __builtin_prefetch(ga + 32, 0, 1);
            __builtin_prefetch(gw + 32, 0, 1);
        }
        *(uint4*)la       = t0;
        *(uint4*)(la + 8) = t1;
        *(uint4*)lw       = t2;
        *(uint4*)(lw + 8) = t3;
        __syncthreads();

        v16h af[2], bf[4];
#pragma unroll
        for (int mt = 0; mt < 2; ++mt) {
            const f16* p = &As1[(wm * 32 + mt * 16 + ml) * 40];
            af[mt] = ld_frag(p + hf * 8, p + 16 + hf * 8);
        }
#pragma unroll
        for (int nt = 0; nt < 4; ++nt) {
            const f16* p = &Ws1[(wn * 64 + nt * 16 + ml) * 40];
            bf[nt] = ld_frag(p + hf * 8, p + 16 + hf * 8);
        }
#pragma unroll
        for (int mt = 0; mt < 2; ++mt)
#pragma unroll
            for (int nt = 0; nt < 4; ++nt)
                acc[mt][nt] = wmma_f16(af[mt], bf[nt], acc[mt][nt]);
        __syncthreads();
    }
#endif

#pragma unroll
    for (int mt = 0; mt < 2; ++mt)
#pragma unroll
        for (int nt = 0; nt < 4; ++nt)
#pragma unroll
            for (int v = 0; v < 8; ++v) {
                int r = bm * 128 + wm * 32 + mt * 16 + hf * 8 + v;
                int c = bn * 128 + wn * 64 + nt * 16 + ml;
                C[(size_t)r * N + c] = acc[mt][nt][v];
            }
}

// ---------------------------------------------------------------------------
// 4) RoPE in place on proj output [s, h*128+d] (rotate-half formulation)
// ---------------------------------------------------------------------------
__global__ __launch_bounds__(256)
void rope_kernel(float* __restrict__ buf, const int* __restrict__ pos, int H) {
    int id = blockIdx.x * 256 + threadIdx.x;
    int i = id & 63;
    int t = id >> 6;
    int h = t % H;
    int s = t / H;
    // inv_freq = theta^(-2i/128)
    float inv = __expf(-9.210340371976184f * (float)(2 * i) * (1.0f / 128.0f));
    float fr = (float)pos[s] * inv;
    float sn, cs;
    __sincosf(fr, &sn, &cs);
    size_t base = (size_t)s * (H * 128) + h * 128;
    float x1 = buf[base + i], x2 = buf[base + i + 64];
    buf[base + i]      = x1 * cs - x2 * sn;
    buf[base + i + 64] = x2 * cs + x1 * sn;
}

// ---------------------------------------------------------------------------
// 5) per-token quant of Q rows (128 wide) -> f16 [NH][S][HD]
// ---------------------------------------------------------------------------
__global__ __launch_bounds__(256)
void quant_q_kernel(const float* __restrict__ qf, f16* __restrict__ qh) {
    int tid = threadIdx.x, lane = tid & 31, wave = tid >> 5;
    int rowid = blockIdx.x * 8 + wave;      // = h*S + s
    int h = rowid >> 11;                    // S = 2^11
    int s = rowid & (S - 1);
    const float* src = qf + (size_t)s * HID + h * 128;
    float x[4];
    float m = 0.0f;
#pragma unroll
    for (int i = 0; i < 4; ++i) {
        x[i] = src[lane + 32 * i];
        m = fmaxf(m, fabsf(x[i]));
    }
#pragma unroll
    for (int msk = 16; msk; msk >>= 1) m = fmaxf(m, __shfl_xor(m, msk, 32));
    float sc = fmaxf(m, 1e-8f) / 127.0f;
    f16* dst = qh + (size_t)rowid * 128;
#pragma unroll
    for (int i = 0; i < 4; ++i) dst[lane + 32 * i] = (f16)q127(x[i], sc);
}

// ---------------------------------------------------------------------------
// 6) per-(head,channel) scales for K and V: max |x| over S
// ---------------------------------------------------------------------------
__global__ __launch_bounds__(256)
void kv_scale_kernel(const float* __restrict__ kf, const float* __restrict__ vf,
                     float* __restrict__ scales) {
    __shared__ float red[256];
    int c = blockIdx.x;                     // 0..2047 (k then v)
    const float* buf = (c < KVW) ? kf : vf;
    int col = c & (KVW - 1);
    int tid = threadIdx.x;
    float m = 0.0f;
    for (int s = tid; s < S; s += 256)
        m = fmaxf(m, fabsf(buf[(size_t)s * KVW + col]));
    red[tid] = m; __syncthreads();
    for (int st = 128; st > 0; st >>= 1) {
        if (tid < st) red[tid] = fmaxf(red[tid], red[tid + st]);
        __syncthreads();
    }
    if (tid == 0) scales[c] = fmaxf(red[0], 1e-8f) / 127.0f;
}

__global__ __launch_bounds__(256)
void quant_k_kernel(const float* __restrict__ kf, const float* __restrict__ scales,
                    f16* __restrict__ kh) {
    int id = blockIdx.x * 256 + threadIdx.x;    // S*1024
    int s = id >> 10, c = id & (KVW - 1);
    int kv = c >> 7, d = c & 127;
    kh[((size_t)kv * S + s) * 128 + d] = (f16)q127(kf[(size_t)s * KVW + c], scales[c]);
}

__global__ __launch_bounds__(256)
void quant_vt_kernel(const float* __restrict__ vf, const float* __restrict__ scales,
                     f16* __restrict__ vth) {
    int id = blockIdx.x * 256 + threadIdx.x;
    int s = id >> 10, c = id & (KVW - 1);
    int kv = c >> 7, d = c & 127;
    // transposed: Vt[kv][d][s] so PV B-fragments are contiguous over keys
    vth[((size_t)kv * 128 + d) * S + s] = (f16)q127(vf[(size_t)s * KVW + c], scales[KVW + c]);
}

// ---------------------------------------------------------------------------
// 7) causal flash attention, WMMA f16 for QK^T and PV
//    grid = (S/64, NH), 128 threads = 4 waves, each wave owns 16 q rows
// ---------------------------------------------------------------------------
__global__ __launch_bounds__(128)
void flash_kernel(const f16* __restrict__ Qh, const f16* __restrict__ Kh,
                  const f16* __restrict__ Vth, float* __restrict__ Ctx) {
    __shared__ f16 Qs[64 * 136];
    __shared__ f16 Ks[64 * 136];
    __shared__ f16 Vts[128 * 72];
    __shared__ f16 Ps[4 * 16 * 72];     // wave-private P staging (C->A layout)

    int qb = blockIdx.x, h = blockIdx.y, kv = h >> 2;   // GQA groups = 4
    int tid = threadIdx.x, lane = tid & 31, w = tid >> 5;
    int ml = lane & 15, hf = lane >> 4;

    // load Q block (64 x 128)
#pragma unroll
    for (int i = 0; i < 4; ++i) {
        int c = tid + 128 * i;
        int r = c >> 3, seg = c & 7;
        const f16* src = Qh + ((size_t)h * S + qb * 64 + r) * 128 + seg * 16;
        f16* dst = &Qs[r * 136 + seg * 16];
#if USE_ASYNC_COPY
        async_cp16(src, dst);
#else
        uint4 t0 = *(const uint4*)src;
        uint4 t1 = *(const uint4*)(src + 8);
        *(uint4*)dst       = t0;
        *(uint4*)(dst + 8) = t1;
#endif
    }

    float m_i[8], l_i[8];
    v8f acc[8];
#pragma unroll
    for (int v = 0; v < 8; ++v) { m_i[v] = -3.0e38f; l_i[v] = 0.0f; }
#pragma unroll
    for (int nt = 0; nt < 8; ++nt)
#pragma unroll
        for (int v = 0; v < 8; ++v) acc[nt][v] = 0.0f;

    f16* pw = &Ps[w * 16 * 72];

    for (int kb = 0; kb <= qb; ++kb) {
        __syncthreads();
        // K block (64 x 128) row-major  +  V^T block (128 d-rows x 64 keys)
#pragma unroll
        for (int i = 0; i < 4; ++i) {
            int c = tid + 128 * i;
            int rk = c >> 3, sgk = c & 7;
            const f16* ksrc = Kh + ((size_t)kv * S + kb * 64 + rk) * 128 + sgk * 16;
            f16* kdst = &Ks[rk * 136 + sgk * 16];
            int rv = c >> 2, sgv = c & 3;
            const f16* vsrc = Vth + ((size_t)kv * 128 + rv) * S + kb * 64 + sgv * 16;
            f16* vdst = &Vts[rv * 72 + sgv * 16];
#if USE_ASYNC_COPY
            async_cp16(ksrc, kdst);
            async_cp16(vsrc, vdst);
#else
            uint4 t0 = *(const uint4*)ksrc;
            uint4 t1 = *(const uint4*)(ksrc + 8);
            uint4 t2 = *(const uint4*)vsrc;
            uint4 t3 = *(const uint4*)(vsrc + 8);
            *(uint4*)kdst       = t0;
            *(uint4*)(kdst + 8) = t1;
            *(uint4*)vdst       = t2;
            *(uint4*)(vdst + 8) = t3;
#endif
        }
#if USE_ASYNC_COPY
        __builtin_amdgcn_s_wait_asynccnt(0);
#endif
        __syncthreads();

        // S = Q K^T : 4 key sub-tiles x 4 d-steps
        v8f sa[4];
#pragma unroll
        for (int nt = 0; nt < 4; ++nt)
#pragma unroll
            for (int v = 0; v < 8; ++v) sa[nt][v] = 0.0f;
#pragma unroll
        for (int ks = 0; ks < 4; ++ks) {
            const f16* ap = &Qs[(w * 16 + ml) * 136 + ks * 32];
            v16h af = ld_frag(ap + hf * 8, ap + 16 + hf * 8);
#pragma unroll
            for (int nt = 0; nt < 4; ++nt) {
                const f16* bp = &Ks[(nt * 16 + ml) * 136 + ks * 32];
                sa[nt] = wmma_f16(af, ld_frag(bp + hf * 8, bp + 16 + hf * 8), sa[nt]);
            }
        }

        // scale + causal mask + online softmax (rows striped hf*8+v, cols = ml)
        int qrow0 = qb * 64 + w * 16 + hf * 8;
        float alpha[8];
#pragma unroll
        for (int v = 0; v < 8; ++v) {
            float mx = -3.0e38f;
#pragma unroll
            for (int nt = 0; nt < 4; ++nt) {
                float sv = sa[nt][v] * SM_SCALE;
                int kg = kb * 64 + nt * 16 + ml;
                if (kg > qrow0 + v) sv = -1.0e9f;
                sa[nt][v] = sv;
                mx = fmaxf(mx, sv);
            }
            mx = fmaxf(mx, __shfl_xor(mx, 1, 32));
            mx = fmaxf(mx, __shfl_xor(mx, 2, 32));
            mx = fmaxf(mx, __shfl_xor(mx, 4, 32));
            mx = fmaxf(mx, __shfl_xor(mx, 8, 32));
            float nm = fmaxf(m_i[v], mx);
            alpha[v] = __expf(m_i[v] - nm);
            m_i[v] = nm;
            float rs = 0.0f;
#pragma unroll
            for (int nt = 0; nt < 4; ++nt) {
                float p = __expf(sa[nt][v] - nm);
                rs += p;
                // int8 fake-quant of probabilities (grid 1/127 of row max)
                sa[nt][v] = rintf(p * 127.0f) * (1.0f / 127.0f);
            }
            rs += __shfl_xor(rs, 1, 32);
            rs += __shfl_xor(rs, 2, 32);
            rs += __shfl_xor(rs, 4, 32);
            rs += __shfl_xor(rs, 8, 32);
            l_i[v] = l_i[v] * alpha[v] + rs;
#pragma unroll
            for (int nt = 0; nt < 8; ++nt) acc[nt][v] *= alpha[v];
        }

        // stage P (C layout -> LDS -> A layout), wave-private
#pragma unroll
        for (int v = 0; v < 8; ++v)
#pragma unroll
            for (int nt = 0; nt < 4; ++nt)
                pw[(hf * 8 + v) * 72 + nt * 16 + ml] = (f16)sa[nt][v];

        // acc += P * V  (2 key-steps x 8 d-tiles)
#pragma unroll
        for (int kstep = 0; kstep < 2; ++kstep) {
            const f16* ap = &pw[ml * 72 + kstep * 32];
            v16h af = ld_frag(ap + hf * 8, ap + 16 + hf * 8);
#pragma unroll
            for (int nt = 0; nt < 8; ++nt) {
                const f16* bp = &Vts[(nt * 16 + ml) * 72 + kstep * 32];
                acc[nt] = wmma_f16(af, ld_frag(bp + hf * 8, bp + 16 + hf * 8), acc[nt]);
            }
        }
    }

    // normalize and write context [h][s][d]
#pragma unroll
    for (int nt = 0; nt < 8; ++nt)
#pragma unroll
        for (int v = 0; v < 8; ++v) {
            float o = acc[nt][v] / l_i[v];
            Ctx[((size_t)h * S + qb * 64 + w * 16 + hf * 8 + v) * 128 + nt * 16 + ml] = o;
        }
}

// ---------------------------------------------------------------------------
// 8) fq_per_token(out) over D=128, transpose to [s, hid], * o_scale2
// ---------------------------------------------------------------------------
__global__ __launch_bounds__(256)
void ctx_quant_kernel(const float* __restrict__ ctx, const float* __restrict__ osc,
                      float* __restrict__ xbuf) {
    int tid = threadIdx.x, lane = tid & 31, wave = tid >> 5;
    int rowid = blockIdx.x * 8 + wave;      // h*S + s
    int h = rowid >> 11;
    int s = rowid & (S - 1);
    const float* src = ctx + (size_t)rowid * 128;
    float x[4];
    float m = 0.0f;
#pragma unroll
    for (int i = 0; i < 4; ++i) {
        x[i] = src[lane + 32 * i];
        m = fmaxf(m, fabsf(x[i]));
    }
#pragma unroll
    for (int msk = 16; msk; msk >>= 1) m = fmaxf(m, __shfl_xor(m, msk, 32));
    float sc = fmaxf(m, 1e-8f) / 127.0f;
    float* dst = xbuf + (size_t)s * HID + h * 128;
#pragma unroll
    for (int i = 0; i < 4; ++i) {
        int d = lane + 32 * i;
        dst[d] = q127(x[i], sc) * osc[h * 128 + d];
    }
}

// ---------------------------------------------------------------------------
// 9) x2 = fq_per_token(x - qbias2) + qbias2  -> f16 [S, HID]
// ---------------------------------------------------------------------------
__global__ __launch_bounds__(256)
void final_quant_kernel(const float* __restrict__ xbuf, const float* __restrict__ qb2,
                        f16* __restrict__ x2h) {
    __shared__ float red[256];
    int s = blockIdx.x, tid = threadIdx.x;
    const float* row = xbuf + (size_t)s * HID;
    float xv[16];
    float m = 0.0f;
#pragma unroll
    for (int i = 0; i < 16; ++i) {
        int j = tid + 256 * i;
        float x = row[j] - qb2[j];
        xv[i] = x;
        m = fmaxf(m, fabsf(x));
    }
    red[tid] = m; __syncthreads();
    for (int st = 128; st > 0; st >>= 1) {
        if (tid < st) red[tid] = fmaxf(red[tid], red[tid + st]);
        __syncthreads();
    }
    float sc = fmaxf(red[0], 1e-8f) / 127.0f;
    f16* orow = x2h + (size_t)s * HID;
#pragma unroll
    for (int i = 0; i < 16; ++i) {
        int j = tid + 256 * i;
        orow[j] = (f16)(q127(xv[i], sc) + qb2[j]);
    }
}

// ---------------------------------------------------------------------------
// launcher
// ---------------------------------------------------------------------------
extern "C" void kernel_launch(void* const* d_in, const int* in_sizes, int n_in,
                              void* d_out, int out_size, void* d_ws, size_t ws_size,
                              hipStream_t stream) {
    (void)in_sizes; (void)n_in; (void)out_size; (void)ws_size;
    const float* hidden = (const float*)d_in[0];
    const int*   pos    = (const int*)d_in[1];
    const float* Wq     = (const float*)d_in[2];
    const float* Wk     = (const float*)d_in[3];
    const float* Wv     = (const float*)d_in[4];
    const float* Wo     = (const float*)d_in[5];
    const float* qb1    = (const float*)d_in[6];
    const float* qb2    = (const float*)d_in[7];
    const float* osc    = (const float*)d_in[8];
    float* out = (float*)d_out;

    char* ws = (char*)d_ws;
    size_t off = 0;
    auto alloc = [&](size_t bytes) -> void* {
        void* p = ws + off;
        off += (bytes + 255) & ~(size_t)255;
        return p;
    };

    f16*   hs_h   = (f16*)alloc((size_t)S * HID * 2);
    f16*   wq_h   = (f16*)alloc((size_t)HID * HID * 2);
    f16*   wk_h   = (f16*)alloc((size_t)KVW * HID * 2);
    f16*   wv_h   = (f16*)alloc((size_t)KVW * HID * 2);
    f16*   wo_h   = (f16*)alloc((size_t)HID * HID * 2);
    float* qf     = (float*)alloc((size_t)S * HID * 4);
    float* kf     = (float*)alloc((size_t)S * KVW * 4);
    float* vf     = (float*)alloc((size_t)S * KVW * 4);
    f16*   qh     = (f16*)alloc((size_t)NH * S * HD * 2);
    f16*   kh     = (f16*)alloc((size_t)NKV * S * HD * 2);
    f16*   vth    = (f16*)alloc((size_t)NKV * HD * S * 2);
    float* scales = (float*)alloc((size_t)2 * KVW * 4);
    float* ctx    = (float*)alloc((size_t)NH * S * HD * 4);
    float* xbuf   = (float*)alloc((size_t)S * HID * 4);
    f16*   x2h    = (f16*)alloc((size_t)S * HID * 2);

    // activation quant
    hs_prep_kernel<<<S, 256, 0, stream>>>(hidden, qb1, hs_h);
    // weights -> f16
    f32_to_f16_kernel<<<(HID * HID + 255) / 256, 256, 0, stream>>>(Wq, wq_h, HID * HID);
    f32_to_f16_kernel<<<(KVW * HID + 255) / 256, 256, 0, stream>>>(Wk, wk_h, KVW * HID);
    f32_to_f16_kernel<<<(KVW * HID + 255) / 256, 256, 0, stream>>>(Wv, wv_h, KVW * HID);
    f32_to_f16_kernel<<<(HID * HID + 255) / 256, 256, 0, stream>>>(Wo, wo_h, HID * HID);
    // projections (WMMA)
    gemm_nt_f16_kernel<<<(S / 128) * (HID / 128), 256, 0, stream>>>(hs_h, wq_h, qf, S, HID, HID);
    gemm_nt_f16_kernel<<<(S / 128) * (KVW / 128), 256, 0, stream>>>(hs_h, wk_h, kf, S, KVW, HID);
    gemm_nt_f16_kernel<<<(S / 128) * (KVW / 128), 256, 0, stream>>>(hs_h, wv_h, vf, S, KVW, HID);
    // RoPE
    rope_kernel<<<(S * NH * 64) / 256, 256, 0, stream>>>(qf, pos, NH);
    rope_kernel<<<(S * NKV * 64) / 256, 256, 0, stream>>>(kf, pos, NKV);
    // quantization of Q / K / V
    quant_q_kernel<<<NH * S / 8, 256, 0, stream>>>(qf, qh);
    kv_scale_kernel<<<2 * KVW, 256, 0, stream>>>(kf, vf, scales);
    quant_k_kernel<<<S * KVW / 256, 256, 0, stream>>>(kf, scales, kh);
    quant_vt_kernel<<<S * KVW / 256, 256, 0, stream>>>(vf, scales, vth);
    // attention (WMMA)
    flash_kernel<<<dim3(S / 64, NH), 128, 0, stream>>>(qh, kh, vth, ctx);
    // epilogue + output projection (WMMA)
    ctx_quant_kernel<<<NH * S / 8, 256, 0, stream>>>(ctx, osc, xbuf);
    final_quant_kernel<<<S, 256, 0, stream>>>(xbuf, qb2, x2h);
    gemm_nt_f16_kernel<<<(S / 128) * (HID / 128), 256, 0, stream>>>(x2h, wo_h, out, S, HID, HID);
}
